// BNN_KWS_75007308857787
// MI455X (gfx1250) — matrix operand order repaired and checked by reading, not compiled, not run on hardware
//
#include <hip/hip_runtime.h>
#include <hip/hip_bf16.h>
#include <stdint.h>

// ---------------------------------------------------------------------------
// CDNA5 (gfx1250) WMMA types and helpers
// ---------------------------------------------------------------------------
typedef __attribute__((ext_vector_type(16))) _Float16 v16h;
typedef __attribute__((ext_vector_type(8)))  float    v8f;

union Frag16 {
    v16h     v;
    _Float16 h[16];
    uint4    q[2];  // 2 x 16B vector loads per fragment
};

// D = A(16x32 f16) * B(32x16 f16) + C(16x16 f32)
__device__ __forceinline__ v8f wmma_f16(v16h a, v16h b, v8f c) {
    return __builtin_amdgcn_wmma_f32_16x16x32_f16(
        /*neg_a=*/false, a, /*neg_b=*/false, b,
        /*c_mod=*/(short)0, c, /*reuse_a=*/false, /*reuse_b=*/false);
}

__device__ __forceinline__ float sgnf(float v) {
    return (v > 0.f) ? 1.f : ((v < 0.f) ? -1.f : 0.f);
}

// ---------------------------------------------------------------------------
// Utility: zero fill (BN stat accumulators and padded-activation halos)
// ---------------------------------------------------------------------------
__global__ void fill_zero_u32(uint32_t* __restrict__ p, long n) {
    long i = (long)blockIdx.x * blockDim.x + threadIdx.x;
    if (i < n) p[i] = 0u;
}

// ---------------------------------------------------------------------------
// conv1: 1->32 channels, 3x3 pad 1, direct (K=9 too small for WMMA).
// Writes pre-BN output in NHWC f16 and accumulates per-channel sum/sumsq.
// ---------------------------------------------------------------------------
__global__ void conv1_bn_stats(const float* __restrict__ x,
                               const float* __restrict__ w,   // [32][1][3][3]
                               _Float16* __restrict__ out,    // [n][80][80][32]
                               float* __restrict__ ssum, float* __restrict__ ssqs)
{
    __shared__ float wsh[32 * 9];
    __shared__ float lsum[32], lsqs[32];
    int t = threadIdx.x;
    for (int i = t; i < 32 * 9; i += blockDim.x) wsh[i] = w[i];
    if (t < 32) { lsum[t] = 0.f; lsqs[t] = 0.f; }
    __syncthreads();

    long gp = (long)blockIdx.x * 256 + t;          // pixel over N*80*80
    int n  = (int)(gp / 6400);
    int p  = (int)(gp % 6400);
    int y  = p / 80, xq = p % 80;
    const float* xin = x + (long)n * 6400;

    float tap[9];
#pragma unroll
    for (int kh = 0; kh < 3; ++kh)
#pragma unroll
        for (int kw = 0; kw < 3; ++kw) {
            int yy = y + kh - 1, xx = xq + kw - 1;
            tap[kh * 3 + kw] =
                (yy >= 0 && yy < 80 && xx >= 0 && xx < 80) ? xin[yy * 80 + xx] : 0.f;
        }

    _Float16* op = out + gp * 32;
    for (int c = 0; c < 32; ++c) {
        float acc = 0.f;
#pragma unroll
        for (int k = 0; k < 9; ++k) acc = fmaf(wsh[c * 9 + k], tap[k], acc);
        op[c] = (_Float16)acc;
        atomicAdd(&lsum[c], acc);
        atomicAdd(&lsqs[c], acc * acc);
    }
    __syncthreads();
    if (t < 32) { atomicAdd(&ssum[t], lsum[t]); atomicAdd(&ssqs[t], lsqs[t]); }
}

// ---------------------------------------------------------------------------
// BN finalize: a = g*rsqrt(var+eps), b' = b - mean*a  (biased variance)
// ---------------------------------------------------------------------------
__global__ void bn_finalize(const float* __restrict__ sum, const float* __restrict__ sqs,
                            const float* __restrict__ g, const float* __restrict__ b,
                            float* __restrict__ a_out, float* __restrict__ b_out,
                            int C, float inv_count)
{
    int c = blockIdx.x * blockDim.x + threadIdx.x;
    if (c >= C) return;
    float m   = sum[c] * inv_count;
    float var = sqs[c] * inv_count - m * m;
    float inv = rsqrtf(var + 1e-5f);
    float a   = g[c] * inv;
    a_out[c]  = a;
    b_out[c]  = b[c] - m * a;
}

// ---------------------------------------------------------------------------
// Fused BN + (ReLU | sign) + 2x2 maxpool into zero-haloed padded NHWC f16.
// in:  [n][2*HO][2*HO][C]   out: [n][HO+2][HO+2][C] (interior at +1)
// ---------------------------------------------------------------------------
template<int SIGN>
__global__ void bn_pool_pad(const _Float16* __restrict__ in,
                            const float* __restrict__ a, const float* __restrict__ bb,
                            _Float16* __restrict__ out, int N, int C, int HO)
{
    long i = (long)blockIdx.x * blockDim.x + threadIdx.x;
    long total = (long)N * HO * HO * C;
    if (i >= total) return;
    int c = (int)(i % C); long r = i / C;
    int ox = (int)(r % HO); r /= HO;
    int oy = (int)(r % HO);
    int n  = (int)(r / HO);
    int HI = 2 * HO, WP = HO + 2;
    const _Float16* p0 = in + (((long)n * HI + 2 * oy) * HI + 2 * ox) * C + c;
    float av = a[c], bv = bb[c];
    float v00 = fmaf(av, (float)p0[0], bv);
    float v01 = fmaf(av, (float)p0[C], bv);
    float v10 = fmaf(av, (float)p0[(long)HI * C], bv);
    float v11 = fmaf(av, (float)p0[(long)HI * C + C], bv);
    float m = fmaxf(fmaxf(v00, v01), fmaxf(v10, v11));
    float o = SIGN ? sgnf(m) : fmaxf(m, 0.f);
    out[(((long)n * WP + (oy + 1)) * WP + (ox + 1)) * C + c] = (_Float16)o;
}

// ---------------------------------------------------------------------------
// Weight binarization: alpha = mean|w| per output row
// ---------------------------------------------------------------------------
__global__ void alpha_reduce(const float* __restrict__ w, float* __restrict__ alpha,
                             int O, int Kl)
{
    __shared__ float red[256];
    int o = blockIdx.x;
    float s = 0.f;
    for (int k = threadIdx.x; k < Kl; k += blockDim.x) s += fabsf(w[(long)o * Kl + k]);
    red[threadIdx.x] = s;
    __syncthreads();
    for (int st = 128; st > 0; st >>= 1) {
        if ((int)threadIdx.x < st) red[threadIdx.x] += red[threadIdx.x + st];
        __syncthreads();
    }
    if (threadIdx.x == 0) alpha[o] = red[0] / (float)Kl;
}

// ---------------------------------------------------------------------------
// Binarize conv weights straight into the WMMA B-fragment layout:
//   wf[oct][tap][chunk][lane][16]  (f16, 32B contiguous per lane per chunk)
// where for a 32x16 B chunk: N = oct*16 + (lane&15), K(ci) = 32*chunk +
// 16*(lane>>4) + j.  Source w is [o][ci][3][3] f32.
// ---------------------------------------------------------------------------
__global__ void binarize_conv_w_frag(const float* __restrict__ w,
                                     const float* __restrict__ alpha,
                                     _Float16* __restrict__ wf, int O, int CI)
{
    int CHUNKS = CI / 32;
    long total = (long)O * CI * 9;
    long i = (long)blockIdx.x * blockDim.x + threadIdx.x;
    if (i >= total) return;
    int j    = (int)(i & 15);
    int lane = (int)((i >> 4) & 31);
    long r   = i >> 9;
    int ch   = (int)(r % CHUNKS); r /= CHUNKS;
    int tap  = (int)(r % 9);
    int oct  = (int)(r / 9);
    int ci = 32 * ch + 16 * (lane >> 4) + j;
    int o  = oct * 16 + (lane & 15);
    float v = w[((long)o * CI + ci) * 9 + tap];
    wf[i] = (_Float16)(sgnf(v) * alpha[o]);
}

// fc1_w [256][3200] -> fragment layout wf[nt][kc][lane][16]
//   N = nt*16 + (lane&15), K = kc*32 + 16*(lane>>4) + j
__global__ void binarize_fc_w_frag(const float* __restrict__ w,
                                   const float* __restrict__ alpha,
                                   _Float16* __restrict__ wf)
{
    long i = (long)blockIdx.x * blockDim.x + threadIdx.x;
    if (i >= (long)256 * 3200) return;
    int j    = (int)(i & 15);
    int lane = (int)((i >> 4) & 31);
    long r   = i >> 9;
    int kc   = (int)(r % 100);
    int nt   = (int)(r / 100);
    int k = kc * 32 + 16 * (lane >> 4) + j;
    int o = nt * 16 + (lane & 15);
    wf[i] = (_Float16)(sgnf(w[(long)o * 3200 + k]) * alpha[o]);
}

// ---------------------------------------------------------------------------
// Binary conv as implicit-im2col WMMA GEMM.
//  act: padded NHWC f16 [n][HO+2][HO+2][CI] (halo==0 == conv zero-padding)
//  wf : fragment-swizzled weights [oct][9][CI/32][32][16] f16
//  out: [n][HO][HO][OC] f16 pre-BN + per-channel sum/sumsq for batch BN.
// One wave per 16-pixel x 16-outch tile; A via 2x global_load_b128, B via
// 2x ds_load_b128 from the LDS-staged per-octile slice.
// ---------------------------------------------------------------------------
template<int CI, int OC, int HO, int WAVES>
__global__ void binconv_wmma(const _Float16* __restrict__ act,
                             const _Float16* __restrict__ wf,
                             _Float16* __restrict__ out,
                             float* __restrict__ ssum, float* __restrict__ ssqs)
{
    constexpr int HP     = HO + 2;
    constexpr int STRIPS = (HO * HO) / 16;
    constexpr int BPI    = STRIPS / WAVES;
    constexpr int NOCT   = OC / 16;
    constexpr int CHUNKS = CI / 32;

    __shared__ uint4 lwq[9 * CHUNKS * 64];   // per-octile B fragments (b128 granular)

    int wave = threadIdx.x >> 5;
    int lane = threadIdx.x & 31;
    int bid  = blockIdx.x;
    int oct  = bid % NOCT;  int r = bid / NOCT;
    int sblk = r % BPI;     int n = r / BPI;
    int oc0  = oct * 16;

    const uint4* gq = (const uint4*)(wf + (long)oct * 9 * CHUNKS * 512);
    for (int idx = threadIdx.x; idx < 9 * CHUNKS * 64; idx += blockDim.x)
        lwq[idx] = gq[idx];
    __syncthreads();

    int strip = sblk * WAVES + wave;
    int M    = lane & 15;
    int half = lane >> 4;
    int Ncol = lane & 15;
    int p = strip * 16 + M;
    int y = p / HO, xq = p % HO;
    const _Float16* pixbase = act + (((long)n * HP + y) * HP + xq) * CI;
    int k0 = 16 * half;  // A: lane holds K = k0..k0+15

    v8f acc = {};
#pragma unroll
    for (int tap = 0; tap < 9; ++tap) {
        int kh = tap / 3, kw = tap % 3;
        const _Float16* prow = pixbase + ((long)kh * HP + kw) * CI;
#pragma unroll
        for (int ch = 0; ch < CHUNKS; ++ch) {
            Frag16 A, B;
            const _Float16* ap = prow + 32 * ch;
            __builtin_prefetch(ap + CI, 0, 0);          // global_prefetch_b8
            A.q[0] = *(const uint4*)(ap + k0);
            A.q[1] = *(const uint4*)(ap + k0 + 8);
            const uint4* bq = &lwq[((tap * CHUNKS + ch) * 32 + lane) * 2];
            B.q[0] = bq[0];
            B.q[1] = bq[1];
            acc = wmma_f16(A.v, B.v, acc);
        }
    }

    // D layout: VGPR r holds D[M = r + 8*half][N = Ncol]
    float lsum = 0.f, lsq = 0.f;
#pragma unroll
    for (int rr = 0; rr < 8; ++rr) {
        int pp = strip * 16 + rr + 8 * half;
        int yy = pp / HO, xx = pp % HO;
        float v = acc[rr];
        out[(((long)n * HO + yy) * HO + xx) * OC + oc0 + Ncol] = (_Float16)v;
        lsum += v; lsq += v * v;
    }
    atomicAdd(&ssum[oc0 + Ncol], lsum);
    atomicAdd(&ssqs[oc0 + Ncol], lsq);
}

// ---------------------------------------------------------------------------
// conv3 epilogue: BN + sign + 2x2 maxpool (20->10) + exact 2x2 avgpool (10->5)
// into hf[n][c*25 + y5*5 + x5] (matches the NCHW reshape order).
// ---------------------------------------------------------------------------
__global__ void bn_sign_pool_avg(const _Float16* __restrict__ in,  // [n][20][20][128]
                                 const float* __restrict__ a, const float* __restrict__ bb,
                                 _Float16* __restrict__ hf)        // [n][3200]
{
    long i = (long)blockIdx.x * blockDim.x + threadIdx.x;
    if (i >= (long)256 * 128 * 25) return;
    int xy = (int)(i % 25); long r = i / 25;
    int c = (int)(r % 128);
    int n = (int)(r / 128);
    int y5 = xy / 5, x5 = xy % 5;
    float av = a[c], bv = bb[c];
    float s = 0.f;
#pragma unroll
    for (int dy = 0; dy < 2; ++dy)
#pragma unroll
        for (int dx = 0; dx < 2; ++dx) {
            int oy = y5 * 2 + dy, ox = x5 * 2 + dx;
            const _Float16* p0 = in + (((long)n * 20 + oy * 2) * 20 + ox * 2) * 128 + c;
            float v00 = fmaf(av, (float)p0[0], bv);
            float v01 = fmaf(av, (float)p0[128], bv);
            float v10 = fmaf(av, (float)p0[20 * 128], bv);
            float v11 = fmaf(av, (float)p0[20 * 128 + 128], bv);
            s += sgnf(fmaxf(fmaxf(v00, v01), fmaxf(v10, v11)));
        }
    hf[(long)n * 3200 + c * 25 + xy] = (_Float16)(0.25f * s);
}

// ---------------------------------------------------------------------------
// fc1: [256 x 3200] @ [3200 x 256] WMMA GEMM -> f32 pre-BN + batch stats.
// One wave per 16x16 tile; B fragments pre-swizzled -> 2x global_load_b128.
// ---------------------------------------------------------------------------
__global__ void fc1_wmma(const _Float16* __restrict__ A,   // [256][3200]
                         const _Float16* __restrict__ Bw,  // frag layout [nt][kc][32][16]
                         float* __restrict__ out,          // [256][256]
                         float* __restrict__ ssum, float* __restrict__ ssqs)
{
    int wave = threadIdx.x >> 5, lane = threadIdx.x & 31;
    int tile = blockIdx.x * 4 + wave;
    int mt = tile >> 4, nt = tile & 15;
    int M = lane & 15, half = lane >> 4, Ncol = lane & 15;
    const _Float16* arow = A + (long)(mt * 16 + M) * 3200;
    const uint4* bbase = (const uint4*)(Bw + (long)nt * 100 * 512) + lane * 2;
    int k0 = 16 * half;

    v8f acc = {};
    for (int kc = 0; kc < 100; ++kc) {
        Frag16 Af, Bf;
        const _Float16* ap = arow + kc * 32;
        if (kc + 1 < 100) __builtin_prefetch(ap + 32, 0, 0);  // global_prefetch_b8
        Af.q[0] = *(const uint4*)(ap + k0);
        Af.q[1] = *(const uint4*)(ap + k0 + 8);
        const uint4* bq = bbase + (long)kc * 64;
        Bf.q[0] = bq[0];
        Bf.q[1] = bq[1];
        acc = wmma_f16(Af.v, Bf.v, acc);
    }

    float lsum = 0.f, lsq = 0.f;
#pragma unroll
    for (int rr = 0; rr < 8; ++rr) {
        int Mrow = rr + 8 * half;
        float v = acc[rr];
        out[(long)(mt * 16 + Mrow) * 256 + nt * 16 + Ncol] = v;
        lsum += v; lsq += v * v;
    }
    atomicAdd(&ssum[nt * 16 + Ncol], lsum);
    atomicAdd(&ssqs[nt * 16 + Ncol], lsq);
}

// BN1d + sign over [256][256]
__global__ void bn_sign_fc(const float* __restrict__ in,
                           const float* __restrict__ a, const float* __restrict__ bb,
                           _Float16* __restrict__ hs)
{
    int i = blockIdx.x * blockDim.x + threadIdx.x;
    if (i >= 256 * 256) return;
    int c = i & 255;
    float v = fmaf(a[c], in[i], bb[c]);
    hs[i] = (_Float16)sgnf(v);
}

// Final head: [256][256] @ fco_w.T [256][12] + bias -> f32 output
__global__ void fco_kernel(const _Float16* __restrict__ hs,
                           const float* __restrict__ w,     // [12][256]
                           const float* __restrict__ bias,
                           float* __restrict__ out)         // [256][12]
{
    int i = blockIdx.x * blockDim.x + threadIdx.x;
    if (i >= 256 * 12) return;
    int j = i % 12, n = i / 12;
    const _Float16* h = hs + (long)n * 256;
    const float* wr = w + (long)j * 256;
    float acc = bias[j];
    for (int k = 0; k < 256; ++k) acc = fmaf((float)h[k], wr[k], acc);
    out[i] = acc;
}

// ---------------------------------------------------------------------------
// Host: workspace carving (sequential region aliasing, ~136 MB peak) and the
// full launch sequence on `stream`.
// ---------------------------------------------------------------------------
extern "C" void kernel_launch(void* const* d_in, const int* in_sizes, int n_in,
                              void* d_out, int out_size, void* d_ws, size_t ws_size,
                              hipStream_t stream)
{
    const float* x       = (const float*)d_in[0];
    const float* conv1_w = (const float*)d_in[1];
    const float* bn1_g   = (const float*)d_in[2];
    const float* bn1_b   = (const float*)d_in[3];
    const float* w2      = (const float*)d_in[4];
    const float* bn2_g   = (const float*)d_in[5];
    const float* bn2_b   = (const float*)d_in[6];
    const float* w3      = (const float*)d_in[7];
    const float* bn3_g   = (const float*)d_in[8];
    const float* bn3_b   = (const float*)d_in[9];
    const float* fc1_w   = (const float*)d_in[10];
    const float* bnf_g   = (const float*)d_in[11];
    const float* bnf_b   = (const float*)d_in[12];
    const float* fco_w   = (const float*)d_in[13];
    const float* fco_b   = (const float*)d_in[14];
    float* out = (float*)d_out;
    (void)in_sizes; (void)n_in; (void)out_size; (void)ws_size;

    char* base = (char*)d_ws;
    const size_t R0 = 104857600;  // conv1out(105MB) / conv2out(52MB) / conv3out(26MB)
    const size_t R1 = 28901376;   // act1(29MB) / act2(16MB) / hfeat(1.6MB)
    _Float16* conv1out = (_Float16*)base;
    _Float16* conv2out = (_Float16*)base;
    _Float16* conv3out = (_Float16*)base;
    _Float16* act1  = (_Float16*)(base + R0);
    _Float16* act2  = (_Float16*)(base + R0);
    _Float16* hfeat = (_Float16*)(base + R0);
    char* s = base + R0 + R1;
    _Float16* we2   = (_Float16*)s;  s += 36864;    // frag [4][9][1][32][16]
    _Float16* we3   = (_Float16*)s;  s += 147456;   // frag [8][9][2][32][16]
    _Float16* wfc1b = (_Float16*)s;  s += 1638400;  // frag [16][100][32][16]
    float*    fc1out= (float*)s;     s += 262144;   // [256][256] f32
    _Float16* hsign = (_Float16*)s;  s += 131072;   // [256][256]
    float*    alpha2= (float*)s;     s += 256;
    float*    alpha3= (float*)s;     s += 512;
    float*    alphaf= (float*)s;     s += 1024;
    float*    stats = (float*)s;     s += 8192;
    float *s1sum = stats,       *s1sqs = stats + 32;
    float *s2sum = stats + 64,  *s2sqs = stats + 128;
    float *s3sum = stats + 192, *s3sqs = stats + 320;
    float *sfsum = stats + 448, *sfsqs = stats + 704;
    float *a1 = stats + 960,  *b1 = stats + 992;
    float *a2 = stats + 1024, *b2 = stats + 1088;
    float *a3 = stats + 1152, *b3 = stats + 1280;
    float *af = stats + 1408, *bf = stats + 1664;

    // 0) zero BN stat accumulators + act1 halo region
    fill_zero_u32<<<4, 256, 0, stream>>>((uint32_t*)stats, 960);
    {
        long n32 = (long)256 * 42 * 42 * 32 / 2;
        fill_zero_u32<<<(int)((n32 + 255) / 256), 256, 0, stream>>>((uint32_t*)act1, n32);
    }

    // 1) conv1 (real) + stats -> BN finalize -> BN/ReLU/pool into padded NHWC
    conv1_bn_stats<<<6400, 256, 0, stream>>>(x, conv1_w, conv1out, s1sum, s1sqs);
    bn_finalize<<<1, 32, 0, stream>>>(s1sum, s1sqs, bn1_g, bn1_b, a1, b1, 32, 1.f / 1638400.f);
    {
        long total = (long)256 * 40 * 40 * 32;
        bn_pool_pad<0><<<(int)((total + 255) / 256), 256, 0, stream>>>(
            conv1out, a1, b1, act1, 256, 32, 40);
    }

    // 2) binarize w2 (fragment swizzle), conv2 via WMMA, BN stats
    alpha_reduce<<<64, 256, 0, stream>>>(w2, alpha2, 64, 288);
    binarize_conv_w_frag<<<(64 * 288 + 255) / 256, 256, 0, stream>>>(w2, alpha2, we2, 64, 32);
    binconv_wmma<32, 64, 40, 4><<<25600, 128, 0, stream>>>(act1, we2, conv2out, s2sum, s2sqs);
    bn_finalize<<<1, 64, 0, stream>>>(s2sum, s2sqs, bn2_g, bn2_b, a2, b2, 64, 1.f / 409600.f);
    {
        long n32 = (long)256 * 22 * 22 * 64 / 2;   // zero act2 halo (act1 now dead)
        fill_zero_u32<<<(int)((n32 + 255) / 256), 256, 0, stream>>>((uint32_t*)act2, n32);
    }
    {
        long total = (long)256 * 20 * 20 * 64;
        bn_pool_pad<1><<<(int)((total + 255) / 256), 256, 0, stream>>>(
            conv2out, a2, b2, act2, 256, 64, 20);
    }

    // 3) binarize w3 (fragment swizzle), conv3 via WMMA, BN stats
    alpha_reduce<<<128, 256, 0, stream>>>(w3, alpha3, 128, 576);
    binarize_conv_w_frag<<<(128 * 576 + 255) / 256, 256, 0, stream>>>(w3, alpha3, we3, 128, 64);
    binconv_wmma<64, 128, 20, 5><<<10240, 160, 0, stream>>>(act2, we3, conv3out, s3sum, s3sqs);
    bn_finalize<<<1, 128, 0, stream>>>(s3sum, s3sqs, bn3_g, bn3_b, a3, b3, 128, 1.f / 102400.f);
    bn_sign_pool_avg<<<3200, 256, 0, stream>>>(conv3out, a3, b3, hfeat);

    // 4) binary fc1 via WMMA GEMM + batch BN + sign
    alpha_reduce<<<256, 256, 0, stream>>>(fc1_w, alphaf, 256, 3200);
    binarize_fc_w_frag<<<(256 * 3200 + 255) / 256, 256, 0, stream>>>(fc1_w, alphaf, wfc1b);
    fc1_wmma<<<64, 128, 0, stream>>>(hfeat, wfc1b, fc1out, sfsum, sfsqs);
    bn_finalize<<<1, 256, 0, stream>>>(sfsum, sfsqs, bnf_g, bnf_b, af, bf, 256, 1.f / 256.f);
    bn_sign_fc<<<256, 256, 0, stream>>>(fc1out, af, bf, hsign);

    // 5) classifier head
    fco_kernel<<<12, 256, 0, stream>>>(hsign, fco_w, fco_b, out);
}
